// rDMinBatch_31731218383282
// MI455X (gfx1250) — matrix operand order repaired and checked by reading, not compiled, not run on hardware
//
#include <hip/hip_runtime.h>

typedef __attribute__((ext_vector_type(16))) _Float16 v16h;
typedef __attribute__((ext_vector_type(8)))  _Float16 v8h;
typedef __attribute__((ext_vector_type(8)))  float    v8f;

#define BB   256   // batch
#define TT   128   // seq len
#define CC   1024  // layer0 input dim
#define HH   256   // hidden per dir
#define G4   1024  // 4*H
#define FMD  512   // 2*H
#define DD   512   // minibatch-disc dim

__device__ __forceinline__ float sigmoidf_(float x) { return 1.f / (1.f + __expf(-x)); }
__device__ __forceinline__ float tanhf_(float x)    { return 1.f - 2.f / (__expf(2.f * x) + 1.f); }

// ---------------------------------------------------------------- converts
__global__ void k_transpose_f16(const float* __restrict__ src, _Float16* __restrict__ dst,
                                int R, int C) {
  int idx = blockIdx.x * 256 + threadIdx.x;
  if (idx >= R * C) return;
  int r = idx / C, c = idx % C;
  dst[(long long)c * R + r] = (_Float16)src[idx];   // dst[K][N] = src[N][K]
}

__global__ void k_convert_f16(const float* __restrict__ src, _Float16* __restrict__ dst, int n) {
  int idx = blockIdx.x * 256 + threadIdx.x;
  if (idx < n) dst[idx] = (_Float16)src[idx];
}

// ------------------------------------------------- f16A x f16W GEMM (f32 acc)
// Out[M,N] = A[M,K] * W[K,N] (+bias).  A row r maps to (t=r>>8, b=r&255):
// element A[r][k] at  A + b*sb + t*st + k   (all f16, strides in elements).
// Block tile 128x128, 8 waves, each wave: 1 M-tile x 8 N-tiles.
__global__ void __launch_bounds__(256) k_gemm_f16(
    const _Float16* __restrict__ A, long long sb, long long st, int K,
    const _Float16* __restrict__ W, const float* __restrict__ bias,
    float* __restrict__ Out, int N) {
  __shared__ __align__(16) _Float16 sBt[128][40];   // [n][k] transposed, padded
  const int tid  = threadIdx.x;
  const int wv   = tid >> 5;
  const int lane = tid & 31;
  const int lmod = lane & 15, lhalf = lane >> 4;
  const int n0 = blockIdx.x * 128;
  const int m0 = blockIdx.y * 128;
  const int mrow = m0 + wv * 16 + lmod;     // global A row for this lane
  const int t_ = mrow >> 8, b_ = mrow & 255;
  const _Float16* arow = A + (long long)b_ * sb + (long long)t_ * st;

  v8f acc[8] = {};

  for (int k0 = 0; k0 < K; k0 += 32) {
    __syncthreads();
    { // stage W[k0..k0+32, n0..n0+128] transposed into LDS
      int r = tid >> 3, cg = tid & 7;
      const _Float16* wp = W + (long long)(k0 + r) * N + n0 + cg * 16;
      _Float16 tmp[16];
#pragma unroll
      for (int e = 0; e < 16; e++) tmp[e] = wp[e];
#pragma unroll
      for (int e = 0; e < 16; e++) sBt[cg * 16 + e][r] = tmp[e];
    }
    __syncthreads();

    v16h afrag;  // A 16x32 f16 fragment per ISA layout (two 16B runs per lane)
    {
      v8h a0 = *(const v8h*)(arow + k0 + lhalf * 8);
      v8h a1 = *(const v8h*)(arow + k0 + 16 + lhalf * 8);
#pragma unroll
      for (int e = 0; e < 8; e++) { afrag[e] = a0[e]; afrag[8 + e] = a1[e]; }
    }
    // preload all 8 B fragments (independent ds clause), then 8 WMMAs b2b
    v16h bfr[8];
#pragma unroll
    for (int nt = 0; nt < 8; nt++) {
      const _Float16* bp = &sBt[nt * 16 + lmod][lhalf * 8];
      v8h bq0 = *(const v8h*)bp, bq1 = *(const v8h*)(bp + 16);
#pragma unroll
      for (int e = 0; e < 8; e++) { bfr[nt][e] = bq0[e]; bfr[nt][8 + e] = bq1[e]; }
    }
#pragma unroll
    for (int nt = 0; nt < 8; nt++)
      acc[nt] = __builtin_amdgcn_wmma_f32_16x16x32_f16(false, afrag, false, bfr[nt],
                                                       (short)0, acc[nt], false, false);
  }

  float bv[8];
  if (bias) {
#pragma unroll
    for (int nt = 0; nt < 8; nt++) bv[nt] = bias[n0 + nt * 16 + lmod];
  } else {
#pragma unroll
    for (int nt = 0; nt < 8; nt++) bv[nt] = 0.f;
  }
#pragma unroll
  for (int nt = 0; nt < 8; nt++) {
    int ncol = n0 + nt * 16 + lmod;
#pragma unroll
    for (int r = 0; r < 8; r++) {
      int row = m0 + wv * 16 + r + lhalf * 8;
      Out[(long long)row * N + ncol] = acc[nt][r] + bv[nt];
    }
  }
}

// ------------------------------------------------------- device-wide barrier
__device__ __forceinline__ void gridBarrier(unsigned* bar, unsigned nblocks) {
  __syncthreads();
  if (threadIdx.x == 0) {
    __threadfence();
    unsigned* cnt = bar;
    unsigned* gen = bar + 16;
    unsigned g = __atomic_load_n(gen, __ATOMIC_ACQUIRE);
    if (atomicAdd(cnt, 1u) == nblocks - 1u) {
      __atomic_store_n(cnt, 0u, __ATOMIC_RELAXED);
      __atomic_fetch_add(gen, 1u, __ATOMIC_RELEASE);
    } else {
      while (__atomic_load_n(gen, __ATOMIC_ACQUIRE) == g) __builtin_amdgcn_s_sleep(2);
    }
    __threadfence();
  }
  __syncthreads();
}

// --------------------------------------------- persistent recurrent LSTM layer
// grid = 32 blocks: blockIdx.x = dir*16 + jblock (16 hidden columns per block).
// Whh columns staged in LDS once; cell state c lives in registers across all t.
// Output sequence written in f16 (feeds next layer's WMMA GEMM directly).
__global__ void __launch_bounds__(256, 1) k_lstm_layer(
    const float* __restrict__ gpre0, const float* __restrict__ gpre1,
    const _Float16* __restrict__ whh0, const _Float16* __restrict__ whh1,  // f16 [256][1024]
    _Float16* __restrict__ hbuf,     // [dir][phase][256*256] f16
    _Float16* __restrict__ outseq,   // [T][B][512] f16
    unsigned* __restrict__ bar) {
  __shared__ __align__(16) _Float16 sW[4][16][264];   // [gate][col][K] padded
  const int tid  = threadIdx.x;
  const int wv   = tid >> 5;
  const int lane = tid & 31;
  const int lmod = lane & 15, lhalf = lane >> 4;
  const int dir = blockIdx.x >> 4;
  const int jb  = blockIdx.x & 15;
  const int j0  = jb * 16;
  const float*    gpre = dir ? gpre1 : gpre0;
  const _Float16* whh  = dir ? whh1 : whh0;
  _Float16* hbase = hbuf + (long long)dir * 2 * 65536;

  { // stage Whh columns for the 4 gates of this block's 16 hidden cols
    int c = tid & 15, g = (tid >> 4) & 3, kr = tid >> 6;
    for (int k = kr; k < 256; k += 4)
      sW[g][c][k] = whh[(long long)k * 1024 + g * 256 + j0 + c];
  }
  { // zero h phase 0 for this dir (each block clears its 4096-half slice)
    _Float16* hz = hbase + jb * 4096;
    for (int i = tid; i < 4096; i += 256) hz[i] = (_Float16)0.f;
  }
  v8f cst[2] = {};
  const int mt0 = wv, mt1 = wv + 8;

  gridBarrier(bar, 32);

  for (int s = 0; s < 128; s++) {
    const int t = dir ? (127 - s) : s;
    const _Float16* hprev = hbase + (long long)(s & 1) * 65536;
    _Float16* hnext = hbase + (long long)((s + 1) & 1) * 65536;

    v8f acc[2][4] = {};
#pragma unroll
    for (int kc = 0; kc < 8; kc++) {
      v16h af[2];
#pragma unroll
      for (int m = 0; m < 2; m++) {
        int mrow = (m ? mt1 : mt0) * 16 + lmod;
        const _Float16* hp = hprev + (long long)mrow * 256 + kc * 32;
        v8h a0 = *(const v8h*)(hp + lhalf * 8);
        v8h a1 = *(const v8h*)(hp + 16 + lhalf * 8);
#pragma unroll
        for (int e = 0; e < 8; e++) { af[m][e] = a0[e]; af[m][8 + e] = a1[e]; }
      }
      // preload all 4 gate B-fragments, then 8 WMMAs back-to-back
      v16h bf[4];
#pragma unroll
      for (int g = 0; g < 4; g++) {
        const _Float16* bp = &sW[g][lmod][kc * 32 + lhalf * 8];
        v8h b0 = *(const v8h*)bp, b1 = *(const v8h*)(bp + 16);
#pragma unroll
        for (int e = 0; e < 8; e++) { bf[g][e] = b0[e]; bf[g][8 + e] = b1[e]; }
      }
#pragma unroll
      for (int g = 0; g < 4; g++) {
        acc[0][g] = __builtin_amdgcn_wmma_f32_16x16x32_f16(false, af[0], false, bf[g],
                                                           (short)0, acc[0][g], false, false);
        acc[1][g] = __builtin_amdgcn_wmma_f32_16x16x32_f16(false, af[1], false, bf[g],
                                                           (short)0, acc[1][g], false, false);
      }
    }
    // elementwise LSTM cell update (gates order: i, f, g, o)
#pragma unroll
    for (int m = 0; m < 2; m++) {
#pragma unroll
      for (int r = 0; r < 8; r++) {
        int row = (m ? mt1 : mt0) * 16 + r + lhalf * 8;   // batch index
        int col = j0 + lmod;                              // hidden index
        long long gb = ((long long)t * 256 + row) * 1024 + col;
        float gi = acc[m][0][r] + gpre[gb];
        float gf = acc[m][1][r] + gpre[gb + 256];
        float gg = acc[m][2][r] + gpre[gb + 512];
        float go = acc[m][3][r] + gpre[gb + 768];
        float cn = sigmoidf_(gf) * cst[m][r] + sigmoidf_(gi) * tanhf_(gg);
        float hn = sigmoidf_(go) * tanhf_(cn);
        cst[m][r] = cn;
        _Float16 h16 = (_Float16)hn;
        hnext[(long long)row * 256 + col] = h16;
        outseq[((long long)t * 256 + row) * 512 + dir * 256 + col] = h16;
      }
    }
    gridBarrier(bar, 32);
  }
}

// ------------------------------------------- minibatch discrimination L1/exp
__global__ void k_minibatch(const float* __restrict__ Ms, float* __restrict__ outT) {
  int d = blockIdx.x * 64 + threadIdx.x;   // grid.x = D/64
  int i = blockIdx.y;                      // grid.y = B
  float mi = Ms[(long long)i * DD + d];
  float s = 0.f;
  for (int j = 0; j < BB; j++) s += __expf(-fabsf(mi - Ms[(long long)j * DD + d]));
  outT[(long long)i * DD + d] = s;
}

// --------------------------------------------------------- final FC + sigmoid
__global__ void k_final(const _Float16* __restrict__ feat, const float* __restrict__ outT,
                        const float* __restrict__ fcw, const float* __restrict__ fcb,
                        float* __restrict__ out) {
  __shared__ float red[256];
  int b = blockIdx.x, tid = threadIdx.x;
  float s = 0.f;
  for (int k = tid; k < FMD; k += 256) s += (float)feat[(long long)b * FMD + k] * fcw[k];
  for (int k = tid; k < DD;  k += 256) s += outT[(long long)b * DD + k] * fcw[FMD + k];
  red[tid] = s;
  __syncthreads();
  for (int off = 128; off > 0; off >>= 1) {
    if (tid < off) red[tid] += red[tid + off];
    __syncthreads();
  }
  if (tid == 0) out[b] = 1.f / (1.f + __expf(-(red[0] + fcb[0])));
}

// ================================================================ launcher
extern "C" void kernel_launch(void* const* d_in, const int* in_sizes, int n_in,
                              void* d_out, int out_size, void* d_ws, size_t ws_size,
                              hipStream_t stream) {
  (void)in_sizes; (void)n_in; (void)out_size; (void)ws_size;
  const float* x      = (const float*)d_in[0];
  const float* wih[3] = {(const float*)d_in[1], (const float*)d_in[4], (const float*)d_in[7]};
  const float* whh[3] = {(const float*)d_in[2], (const float*)d_in[5], (const float*)d_in[8]};
  const float* bl[3]  = {(const float*)d_in[3], (const float*)d_in[6], (const float*)d_in[9]};
  const float* Tt     = (const float*)d_in[10];
  const float* fcw    = (const float*)d_in[11];
  const float* fcb    = (const float*)d_in[12];
  float* out = (float*)d_out;

  // ---- workspace carve-up (256B aligned)
  char* ws = (char*)d_ws;
  size_t off = 0;
  auto carve = [&](size_t bytes) { char* p = ws + off; off = (off + bytes + 255) & ~(size_t)255; return p; };
  unsigned*  bar     = (unsigned*)carve(256);
  const int  inD[3]  = {CC, FMD, FMD};
  _Float16*  wih16[3][2];
  _Float16*  whh16[3][2];
  for (int l = 0; l < 3; l++)
    for (int d = 0; d < 2; d++) {
      wih16[l][d] = (_Float16*)carve((size_t)inD[l] * G4 * 2);
      whh16[l][d] = (_Float16*)carve((size_t)HH * G4 * 2);
    }
  float*    gpre0 = (float*)carve((size_t)TT * BB * G4 * 4);
  float*    gpre1 = (float*)carve((size_t)TT * BB * G4 * 4);
  _Float16* x16   = (_Float16*)carve((size_t)BB * TT * CC * 2);
  _Float16* seq0  = (_Float16*)carve((size_t)TT * BB * FMD * 2);
  _Float16* seq1  = (_Float16*)carve((size_t)TT * BB * FMD * 2);
  _Float16* hbuf  = (_Float16*)carve((size_t)2 * 2 * BB * HH * 2);
  _Float16* tt16  = (_Float16*)carve((size_t)FMD * DD * 2);
  float* Ms   = (float*)carve((size_t)BB * DD * 4);
  float* outT = (float*)carve((size_t)BB * DD * 4);

  hipMemsetAsync(bar, 0, 256, stream);

  // ---- weight + activation conversion (f32 -> f16; weights also transposed)
  for (int l = 0; l < 3; l++)
    for (int d = 0; d < 2; d++) {
      int nih = G4 * inD[l];
      k_transpose_f16<<<(nih + 255) / 256, 256, 0, stream>>>(
          wih[l] + (size_t)d * nih, wih16[l][d], G4, inD[l]);
      int nhh = G4 * HH;
      k_transpose_f16<<<(nhh + 255) / 256, 256, 0, stream>>>(
          whh[l] + (size_t)d * nhh, whh16[l][d], G4, HH);
    }
  k_convert_f16<<<(FMD * DD + 255) / 256, 256, 0, stream>>>(Tt, tt16, FMD * DD);
  {
    int nx = BB * TT * CC;
    k_convert_f16<<<(nx + 255) / 256, 256, 0, stream>>>(x, x16, nx);
  }

  // ---- layers
  const _Float16* A = x16;
  long long sb = (long long)TT * CC, st_ = CC;   // layer0 input is [B,T,C]
  _Float16* seqs[2] = {seq0, seq1};
  for (int l = 0; l < 3; l++) {
    int K = inD[l];
    dim3 gridP(G4 / 128, (TT * BB) / 128);
    k_gemm_f16<<<gridP, 256, 0, stream>>>(A, sb, st_, K, wih16[l][0], bl[l],      gpre0, G4);
    k_gemm_f16<<<gridP, 256, 0, stream>>>(A, sb, st_, K, wih16[l][1], bl[l] + G4, gpre1, G4);
    _Float16* so = seqs[l & 1];
    k_lstm_layer<<<32, 256, 0, stream>>>(gpre0, gpre1, whh16[l][0], whh16[l][1], hbuf, so, bar);
    A = so; sb = FMD; st_ = (long long)BB * FMD;  // next layer input [T,B,2H]
  }

  // ---- feat = last timestep of layer-2 output (f16); Ms = feat @ T_tensor
  const _Float16* feat = seqs[0] + (size_t)(TT - 1) * BB * FMD;   // layer2 wrote seqs[0]
  k_gemm_f16<<<dim3(DD / 128, BB / 128), 256, 0, stream>>>(
      feat, (long long)FMD, 0, FMD, tt16, nullptr, Ms, DD);

  k_minibatch<<<dim3(DD / 64, BB), 64, 0, stream>>>(Ms, outT);
  k_final<<<BB, 256, 0, stream>>>(feat, outT, fcw, fcb, out);
}
